// NLoc_42417097017196
// MI455X (gfx1250) — matrix operand order repaired and checked by reading, not compiled, not run on hardware
//
#include <hip/hip_runtime.h>

// ------------------------------------------------------------------
// ViT encoder forward for MI455X (gfx1250, wave32, WMMA).
// All matmuls via v_wmma_f32_16x16x32_bf16 (fp32 accum).
// Weights pre-transposed to bf16 N x K so all LDS fragment reads are
// contiguous b128; global->LDS tile staging uses CDNA5 async loads
// (global_load_async_to_lds_b128, ASYNCcnt).
// Token dim padded 513 -> 544; padded keys masked in softmax.
// ------------------------------------------------------------------

typedef unsigned int  uint32;
typedef unsigned short ushort16;
typedef __attribute__((ext_vector_type(16))) __bf16 v16bf;
typedef __attribute__((ext_vector_type(8)))  float  v8f;

#define BB    16
#define NTOK  513
#define NPAD  544
#define EDIM  768
#define HH    12
#define DD    64
#define LL    12
#define FF    3072
#define QKV3  2304
#define MROWS (BB*NPAD)      // 8704
#define MEMB  (BB*512)       // 8192
#define INV_SCALE (1.0f/27.712812921102035f)   // 1/sqrt(768)

union BF16x16 { v16bf v; uint32 u[8]; ushort16 s[16]; };

__device__ __forceinline__ ushort16 f2bf(float f) {
  uint32 u = __float_as_uint(f);
  uint32 r = u + 0x7FFFu + ((u >> 16) & 1u);
  return (ushort16)(r >> 16);
}

// CDNA5 async global -> LDS copy, 128-bit, tracked by ASYNCcnt.
__device__ __forceinline__ void async_ld_b128(const void* gaddr, const void* lds) {
  unsigned int l = (unsigned int)(unsigned long long)lds;  // aperture rule: LDS address is the pointer's low 32 bits
  asm volatile("global_load_async_to_lds_b128 %0, %1, off"
               :: "v"(l), "v"((unsigned long long)gaddr)
               : "memory");
}
__device__ __forceinline__ void wait_async0() {
  asm volatile("s_wait_asynccnt 0x0" ::: "memory");
}

// A-matrix 16x32 bf16 fragment (ISA 7.12.2): lane half selects K groups.
__device__ __forceinline__ v16bf load_a16(const ushort16* p, int stride, int lane) {
  BF16x16 r;
  int m   = lane & 15;
  int klo = (lane < 16) ? 0  : 8;
  int khi = (lane < 16) ? 16 : 24;
  const ushort16* row = p + m * stride;
#pragma unroll
  for (int j = 0; j < 4; ++j) r.u[j]     = *(const uint32*)(row + klo + 2*j);
#pragma unroll
  for (int j = 0; j < 4; ++j) r.u[4 + j] = *(const uint32*)(row + khi + 2*j);
  return r.v;
}

// B-matrix 32x16 bf16 fragment from memory holding B^T rows contiguously:
// lanes 0-15 take K=0..7,16..23, lanes 16-31 take K=8..15,24..31, pairs adjacent.
__device__ __forceinline__ v16bf load_bT16(const ushort16* rowp, int lane) {
  BF16x16 r;
  int kb = (lane < 16) ? 0 : 8;
#pragma unroll
  for (int j = 0; j < 8; ++j) r.u[j] = *(const uint32*)(rowp + kb + 2*(j & 3) + ((j >> 2) << 4));
  return r.v;
}

// ------------------------------------------------------------------
// Generic bf16 GEMM: C = A(MxK) @ B(KxN) + bias, B given TRANSPOSED (NxK).
// WG: 256 threads = 8 waves (2x4), tile 64M x 256N, K-step 32.
// MODE 0: embed (fp32 out, row remap b/t, +pos)
// MODE 1: qkv scatter -> Qb, Kb row-major (t,d); Vt transposed (d,t)
// MODE 2: gelu -> bf16 out
// MODE 3: residual fp32 in-place (hout += result)
// ------------------------------------------------------------------
template <int MODE>
__global__ __launch_bounds__(256) void gemm_bf16(
    const ushort16* __restrict__ A, const ushort16* __restrict__ BT,
    const float* __restrict__ bias, int M, int N, int K,
    float* __restrict__ hout, const float* __restrict__ pos,
    ushort16* __restrict__ outb, ushort16* __restrict__ Qb,
    ushort16* __restrict__ Kb, ushort16* __restrict__ Vt) {
  __shared__ __align__(16) ushort16 As[64 * 32];      // 64 rows(M) x 32 k
  __shared__ __align__(16) ushort16 BsT[256 * 32];    // 256 rows(N) x 32 k

  const int tid  = threadIdx.x;
  const int lane = tid & 31;
  const int wave = tid >> 5;
  const int wm = wave >> 2;       // 0..1
  const int wn = wave & 3;        // 0..3
  const size_t blockM = (size_t)blockIdx.y * 64;
  const size_t blockN = (size_t)blockIdx.x * 256;

  v8f acc[2][4];
#pragma unroll
  for (int mi = 0; mi < 2; ++mi)
#pragma unroll
    for (int ni = 0; ni < 4; ++ni) { v8f z = {}; acc[mi][ni] = z; }

  for (int k0 = 0; k0 < K; k0 += 32) {
    // async stage A tile 64x32 : one b128 per thread
    {
      int r = tid >> 2, c = (tid & 3) * 8;
      async_ld_b128(A + (blockM + r) * (size_t)K + k0 + c, As + r * 32 + c);
    }
    // async stage BT tile 256x32 : four b128 per thread (row n = tid)
    {
      const ushort16* src = BT + (blockN + tid) * (size_t)K + k0;
      ushort16* dst = BsT + tid * 32;
#pragma unroll
      for (int j = 0; j < 4; ++j) async_ld_b128(src + j * 8, dst + j * 8);
    }
    wait_async0();
    __syncthreads();

    v16bf af[2];
#pragma unroll
    for (int mi = 0; mi < 2; ++mi)
      af[mi] = load_a16(As + (wm * 32 + mi * 16) * 32, 32, lane);
    v16bf bfm[4];
#pragma unroll
    for (int ni = 0; ni < 4; ++ni)
      bfm[ni] = load_bT16(BsT + (wn * 64 + ni * 16 + (lane & 15)) * 32, lane);

#pragma unroll
    for (int mi = 0; mi < 2; ++mi)
#pragma unroll
      for (int ni = 0; ni < 4; ++ni)
        acc[mi][ni] = __builtin_amdgcn_wmma_f32_16x16x32_bf16(
            false, af[mi], false, bfm[ni], (short)0, acc[mi][ni], false, false);
    __syncthreads();
  }

  // epilogue
#pragma unroll
  for (int mi = 0; mi < 2; ++mi) {
#pragma unroll
    for (int ni = 0; ni < 4; ++ni) {
#pragma unroll
      for (int r = 0; r < 8; ++r) {
        size_t row = blockM + wm * 32 + mi * 16 + ((lane & 16) ? r + 8 : r);
        int col = (int)(blockN + wn * 64 + ni * 16 + (lane & 15));
        float v = acc[mi][ni][r] + bias[col];
        if (MODE == 0) {            // embed: x rows (b, t-1) -> h rows (b, t)
          size_t b = row / 512, t = row % 512 + 1;
          hout[(b * NPAD + t) * EDIM + col] = v + pos[t * (size_t)EDIM + col];
        } else if (MODE == 1) {     // qkv scatter, col = h*192 + d*3 + s
          size_t b = row / NPAD, t = row % NPAD;
          int h = col / 192, rem = col - h * 192;
          int d = rem / 3, s = rem - d * 3;
          ushort16 bv = f2bf(v);
          size_t bh = b * HH + h;
          if (s == 0)      Qb[(bh * NPAD + t) * DD + d]  = bv;
          else if (s == 1) Kb[(bh * NPAD + t) * DD + d]  = bv;
          else             Vt[(bh * DD + d) * NPAD + t]  = bv;   // transposed
        } else if (MODE == 2) {     // exact GELU -> bf16
          float g = 0.5f * v * (1.0f + erff(v * 0.70710678118654752f));
          outb[row * (size_t)N + col] = f2bf(g);
        } else {                    // residual fp32, in-place
          size_t idx = row * (size_t)EDIM + col;
          hout[idx] = hout[idx] + v;
        }
      }
    }
  }
}

// ------------------------------------------------------------------
// Flash-style attention: 1 wave per (qtile, head, batch).
// S = Q K^T (4 WMMA), online softmax (shfl within 16-lane halves),
// O += P V (4 WMMA, P reshaped C->A layout through LDS).
// ------------------------------------------------------------------
__global__ __launch_bounds__(32) void attn_kernel(
    const ushort16* __restrict__ Qb, const ushort16* __restrict__ Kb,
    const ushort16* __restrict__ Vt, ushort16* __restrict__ Ob) {
  __shared__ __align__(16) ushort16 Ps[16 * 32];

  const int lane = threadIdx.x;
  const int qt = blockIdx.x;   // 0..33
  const int h  = blockIdx.y;   // 0..11
  const int b  = blockIdx.z;   // 0..15
  const size_t bh = (size_t)b * HH + h;

  const ushort16* Qbase = Qb + (bh * NPAD + (size_t)qt * 16) * DD;
  const ushort16* Kbase = Kb + bh * NPAD * DD;
  const ushort16* Vbase = Vt + bh * DD * NPAD;

  v16bf aq[2];
#pragma unroll
  for (int dc = 0; dc < 2; ++dc) aq[dc] = load_a16(Qbase + dc * 32, DD, lane);

  v8f o[4];
#pragma unroll
  for (int nf = 0; nf < 4; ++nf) { v8f z = {}; o[nf] = z; }
  float rmax[8], rsum[8];
#pragma unroll
  for (int r = 0; r < 8; ++r) { rmax[r] = -1e30f; rsum[r] = 0.0f; }

  for (int kt = 0; kt < NPAD / 32; ++kt) {        // 17 key tiles of 32
    v8f s[2];
#pragma unroll
    for (int kn = 0; kn < 2; ++kn) {
      v8f z = {}; s[kn] = z;
      int key = kt * 32 + kn * 16 + (lane & 15);
      const ushort16* krow = Kbase + (size_t)key * DD;
#pragma unroll
      for (int dc = 0; dc < 2; ++dc) {
        v16bf bk = load_bT16(krow + dc * 32, lane);
        s[kn] = __builtin_amdgcn_wmma_f32_16x16x32_bf16(
            false, aq[dc], false, bk, (short)0, s[kn], false, false);
      }
      if (key >= NTOK) {
#pragma unroll
        for (int r = 0; r < 8; ++r) s[kn][r] = -1e30f;
      }
    }

    // online softmax per row (8 rows per lane; reductions within 16-lane half)
#pragma unroll
    for (int r = 0; r < 8; ++r) {
      float tmax = fmaxf(s[0][r], s[1][r]);
#pragma unroll
      for (int off = 1; off < 16; off <<= 1) tmax = fmaxf(tmax, __shfl_xor(tmax, off));
      float nm = fmaxf(rmax[r], tmax);
      float sc = __expf(rmax[r] - nm);
      float p0 = __expf(s[0][r] - nm);
      float p1 = __expf(s[1][r] - nm);
      float ps = p0 + p1;
#pragma unroll
      for (int off = 1; off < 16; off <<= 1) ps += __shfl_xor(ps, off);
      rsum[r] = rsum[r] * sc + ps;
      rmax[r] = nm;
#pragma unroll
      for (int nf = 0; nf < 4; ++nf) o[nf][r] = o[nf][r] * sc;
      int m = (lane & 16) ? r + 8 : r;
      Ps[m * 32 + (lane & 15)]      = f2bf(p0);
      Ps[m * 32 + 16 + (lane & 15)] = f2bf(p1);
    }
    __syncthreads();

    v16bf ap = load_a16(Ps, 32, lane);
#pragma unroll
    for (int nf = 0; nf < 4; ++nf) {
      int d = nf * 16 + (lane & 15);
      v16bf bv = load_bT16(Vbase + (size_t)d * NPAD + kt * 32, lane);
      o[nf] = __builtin_amdgcn_wmma_f32_16x16x32_bf16(
          false, ap, false, bv, (short)0, o[nf], false, false);
    }
    __syncthreads();
  }

  // normalize by rowsum, apply post-softmax 1/sqrt(E), write concat-head O
#pragma unroll
  for (int r = 0; r < 8; ++r) {
    float inv = INV_SCALE / rsum[r];
    int m = (lane & 16) ? r + 8 : r;
    size_t t = (size_t)qt * 16 + m;
#pragma unroll
    for (int nf = 0; nf < 4; ++nf) {
      int d = nf * 16 + (lane & 15);
      Ob[((size_t)b * NPAD + t) * EDIM + h * DD + d] = f2bf(o[nf][r] * inv);
    }
  }
}

// ------------------------------------------------------------------
// LayerNorm: one 256-thread block per row, 3 elems/thread (E=768).
// ------------------------------------------------------------------
__global__ __launch_bounds__(256) void ln_kernel(
    const float* __restrict__ h, const float* __restrict__ g,
    const float* __restrict__ be, ushort16* __restrict__ y) {
  __shared__ float red[256];
  __shared__ float s_mu, s_rs;
  const int tid = threadIdx.x;
  const size_t row = blockIdx.x;
  const float* x = h + row * EDIM;
  float v[3], s = 0.0f;
#pragma unroll
  for (int k = 0; k < 3; ++k) { v[k] = x[tid + k * 256]; s += v[k]; }
  red[tid] = s; __syncthreads();
  for (int o = 128; o > 0; o >>= 1) { if (tid < o) red[tid] += red[tid + o]; __syncthreads(); }
  if (tid == 0) s_mu = red[0] * (1.0f / EDIM);
  __syncthreads();
  float mu = s_mu; s = 0.0f;
#pragma unroll
  for (int k = 0; k < 3; ++k) { float d = v[k] - mu; s += d * d; }
  red[tid] = s; __syncthreads();
  for (int o = 128; o > 0; o >>= 1) { if (tid < o) red[tid] += red[tid + o]; __syncthreads(); }
  if (tid == 0) s_rs = rsqrtf(red[0] * (1.0f / EDIM) + 1e-5f);
  __syncthreads();
  float rs = s_rs;
#pragma unroll
  for (int k = 0; k < 3; ++k) {
    int c = tid + k * 256;
    y[row * EDIM + c] = f2bf((v[k] - mu) * rs * g[c] + be[c]);
  }
}

// ------------------------------------------------------------------
// Small helpers
// ------------------------------------------------------------------
__global__ void f2bf_kernel(const float* __restrict__ src, ushort16* __restrict__ dst, long n) {
  for (long i = (long)blockIdx.x * blockDim.x + threadIdx.x; i < n;
       i += (long)gridDim.x * blockDim.x)
    dst[i] = f2bf(src[i]);
}

// transpose-convert: src fp32 (K x N) row-major -> dst bf16 (N x K)
__global__ void f2bfT_kernel(const float* __restrict__ src, ushort16* __restrict__ dst,
                             int K, int N) {
  long total = (long)K * N;
  for (long i = (long)blockIdx.x * blockDim.x + threadIdx.x; i < total;
       i += (long)gridDim.x * blockDim.x) {
    long n = i / K, k = i % K;
    dst[i] = f2bf(src[k * (long)N + n]);
  }
}

__global__ void cls_kernel(float* __restrict__ h, const float* __restrict__ cls,
                           const float* __restrict__ pos) {
  long i = (long)blockIdx.x * blockDim.x + threadIdx.x;
  if (i >= (long)BB * EDIM) return;
  long b = i / EDIM, c = i % EDIM;
  h[(b * NPAD) * (long)EDIM + c] = cls[c] + pos[c];
}

__global__ void pad_kernel(float* __restrict__ h) {
  long n = (long)BB * (NPAD - NTOK) * EDIM;
  for (long i = (long)blockIdx.x * blockDim.x + threadIdx.x; i < n;
       i += (long)gridDim.x * blockDim.x) {
    long per = (long)(NPAD - NTOK) * EDIM;
    long b = i / per, r = i % per;
    long t = NTOK + r / EDIM, c = r % EDIM;
    h[(b * NPAD + t) * (long)EDIM + c] = 0.0f;
  }
}

__global__ void extract_kernel(const float* __restrict__ h, float* __restrict__ out) {
  long i = (long)blockIdx.x * blockDim.x + threadIdx.x;
  if (i >= (long)BB * EDIM) return;
  long b = i / EDIM, c = i % EDIM;
  out[i] = h[(b * NPAD) * (long)EDIM + c];
}

// ------------------------------------------------------------------
// Host orchestration
// ------------------------------------------------------------------
static inline int cblocks(long n) {
  long b = (n + 255) / 256;
  return (int)(b > 2048 ? 2048 : b);
}

extern "C" void kernel_launch(void* const* d_in, const int* in_sizes, int n_in,
                              void* d_out, int out_size, void* d_ws, size_t ws_size,
                              hipStream_t stream) {
  const float* x      = (const float*)d_in[0];
  const float* proj_w = (const float*)d_in[1];
  const float* proj_b = (const float*)d_in[2];
  const float* cls    = (const float*)d_in[3];
  const float* pos    = (const float*)d_in[4];
  const float* ln1_g  = (const float*)d_in[5];
  const float* ln1_b  = (const float*)d_in[6];
  const float* qkv_w  = (const float*)d_in[7];
  const float* qkv_b  = (const float*)d_in[8];
  const float* out_w  = (const float*)d_in[9];
  const float* out_b  = (const float*)d_in[10];
  const float* ln2_g  = (const float*)d_in[11];
  const float* ln2_b  = (const float*)d_in[12];
  const float* ff1_w  = (const float*)d_in[13];
  const float* ff1_b  = (const float*)d_in[14];
  const float* ff2_w  = (const float*)d_in[15];
  const float* ff2_b  = (const float*)d_in[16];

  // workspace layout (16B aligned blocks)
  char* ws = (char*)d_ws;
  size_t off = 0;
  float*    h  = (float*)(ws + off);    off += (size_t)MROWS * EDIM * 4;
  ushort16* y  = (ushort16*)(ws + off); off += (size_t)MROWS * EDIM * 2;   // also x_bf16
  ushort16* g  = (ushort16*)(ws + off); off += (size_t)MROWS * FF * 2;
  ushort16* Qb = (ushort16*)(ws + off); off += (size_t)BB * HH * NPAD * DD * 2;
  ushort16* Kb = (ushort16*)(ws + off); off += (size_t)BB * HH * NPAD * DD * 2;
  ushort16* Vt = (ushort16*)(ws + off); off += (size_t)BB * HH * DD * NPAD * 2;
  ushort16* Ob = (ushort16*)(ws + off); off += (size_t)MROWS * EDIM * 2;
  ushort16* wb = (ushort16*)(ws + off); off += (size_t)EDIM * FF * 2;      // weight staging (NxK)

  // ---- embed: h[:,1:513] = x @ proj_w + b + pos ; h[:,0] = cls+pos ; pad 0
  f2bf_kernel<<<cblocks((long)MEMB * EDIM), 256, 0, stream>>>(x, y, (long)MEMB * EDIM);
  f2bfT_kernel<<<cblocks((long)EDIM * EDIM), 256, 0, stream>>>(proj_w, wb, EDIM, EDIM);
  gemm_bf16<0><<<dim3(EDIM / 256, MEMB / 64), 256, 0, stream>>>(
      y, wb, proj_b, MEMB, EDIM, EDIM, h, pos, nullptr, nullptr, nullptr, nullptr);
  cls_kernel<<<(BB * EDIM + 255) / 256, 256, 0, stream>>>(h, cls, pos);
  pad_kernel<<<cblocks((long)BB * (NPAD - NTOK) * EDIM), 256, 0, stream>>>(h);

  for (int l = 0; l < LL; ++l) {
    // LN1 -> y
    ln_kernel<<<MROWS, 256, 0, stream>>>(h, ln1_g + (size_t)l * EDIM,
                                         ln1_b + (size_t)l * EDIM, y);
    // QKV
    f2bfT_kernel<<<cblocks((long)EDIM * QKV3), 256, 0, stream>>>(
        qkv_w + (size_t)l * EDIM * QKV3, wb, EDIM, QKV3);
    gemm_bf16<1><<<dim3(QKV3 / 256, MROWS / 64), 256, 0, stream>>>(
        y, wb, qkv_b + (size_t)l * QKV3, MROWS, QKV3, EDIM,
        nullptr, nullptr, nullptr, Qb, Kb, Vt);
    // attention
    attn_kernel<<<dim3(NPAD / 16, HH, BB), 32, 0, stream>>>(Qb, Kb, Vt, Ob);
    // out proj + residual
    f2bfT_kernel<<<cblocks((long)EDIM * EDIM), 256, 0, stream>>>(
        out_w + (size_t)l * EDIM * EDIM, wb, EDIM, EDIM);
    gemm_bf16<3><<<dim3(EDIM / 256, MROWS / 64), 256, 0, stream>>>(
        Ob, wb, out_b + (size_t)l * EDIM, MROWS, EDIM, EDIM,
        h, nullptr, nullptr, nullptr, nullptr, nullptr);
    // LN2 -> y
    ln_kernel<<<MROWS, 256, 0, stream>>>(h, ln2_g + (size_t)l * EDIM,
                                         ln2_b + (size_t)l * EDIM, y);
    // FF1 + GELU -> g
    f2bfT_kernel<<<cblocks((long)EDIM * FF), 256, 0, stream>>>(
        ff1_w + (size_t)l * EDIM * FF, wb, EDIM, FF);
    gemm_bf16<2><<<dim3(FF / 256, MROWS / 64), 256, 0, stream>>>(
        y, wb, ff1_b + (size_t)l * FF, MROWS, FF, EDIM,
        nullptr, nullptr, g, nullptr, nullptr, nullptr);
    // FF2 + residual
    f2bfT_kernel<<<cblocks((long)FF * EDIM), 256, 0, stream>>>(
        ff2_w + (size_t)l * FF * EDIM, wb, FF, EDIM);
    gemm_bf16<3><<<dim3(EDIM / 256, MROWS / 64), 256, 0, stream>>>(
        g, wb, ff2_b + (size_t)l * EDIM, MROWS, EDIM, FF,
        h, nullptr, nullptr, nullptr, nullptr, nullptr);
  }

  extract_kernel<<<(BB * EDIM + 255) / 256, 256, 0, stream>>>(h, (float*)d_out);
}
